// MRGCN_75402445849167
// MI455X (gfx1250) — matrix operand order repaired and checked by reading, not compiled, not run on hardware
//
#include <hip/hip_runtime.h>
#include <hip/hip_bf16.h>

// ---------------------------------------------------------------------------
// Problem constants (match reference)
// ---------------------------------------------------------------------------
#define NN   50000
#define EE   500000
#define HH   128
#define LL   3
#define RR   8
#define BB   30
#define GG   16
#define CC   8
#define SLICES 9            // root + 8 relations
#define EPSF 1e-5f

typedef __attribute__((ext_vector_type(16))) __bf16 v16bf;
typedef __attribute__((ext_vector_type(8)))  float  v8f;

union Frag { v16bf v; unsigned u[8]; uint4 q[2]; };

__device__ __forceinline__ float bf2f(unsigned short h) {
    return __uint_as_float(((unsigned)h) << 16);
}
__device__ __forceinline__ unsigned short f2bf(float f) {
    unsigned x = __float_as_uint(f);
    return (unsigned short)((x + 0x7FFFu + ((x >> 16) & 1u)) >> 16);
}

// ---------------------------------------------------------------------------
// 1. BatchNorm1 (eval) + convert to bf16 activations
// ---------------------------------------------------------------------------
__global__ void bn1_kernel(const float* __restrict__ x,
                           const float* __restrict__ g, const float* __restrict__ b,
                           const float* __restrict__ m, const float* __restrict__ var,
                           unsigned short* __restrict__ hb) {
    int i = blockIdx.x * blockDim.x + threadIdx.x;   // < N*H
    int c = i & (HH - 1);
    float v = (x[i] - m[c]) * rsqrtf(var[c] + EPSF) * g[c] + b[c];
    hb[i] = f2bf(v);
}

// ---------------------------------------------------------------------------
// 2. Combine basis weights and emit them DIRECTLY in WMMA B-fragment layout.
//    Each thread produces one packed dword {W[k][n], W[k+1][n]} (k even).
//    Packed index: ((((s*8 + ntile)*4 + kk)*32 + lane)*8 + v)
//      lane = (kgrp<<4) | (n&15), kgrp = ((k&31)>>3)&1, v = ((k&7)>>1) | (((k&31)>>4)<<2), kk = k>>5
// ---------------------------------------------------------------------------
__global__ void wcombine_pack_kernel(const float* __restrict__ basis,
                                     const float* __restrict__ comp,
                                     const float* __restrict__ root,
                                     int l, unsigned* __restrict__ Bpack) {
    int idx = blockIdx.x * blockDim.x + threadIdx.x;   // < 9 * 64 * 128
    int s     = idx >> 13;          // slice 0..8
    int rem   = idx & 8191;
    int kpair = rem >> 7;           // 0..63
    int n     = rem & 127;
    int k     = kpair * 2;

    int e0 = k * HH + n;
    int e1 = (k + 1) * HH + n;
    float w0, w1;
    if (s == 0) {
        const float* rp = root + (size_t)l * HH * HH;
        w0 = rp[e0];
        w1 = rp[e1];
    } else {
        int r = s - 1;
        const float* cp = comp + ((size_t)l * RR + r) * BB;
        float a0 = 0.f, a1 = 0.f;
        #pragma unroll 6
        for (int bb = 0; bb < BB; ++bb) {
            const float* bp = basis + ((size_t)l * BB + bb) * HH * HH;
            float c = cp[bb];
            a0 += c * bp[e0];
            a1 += c * bp[e1];
        }
        w0 = a0; w1 = a1;
    }
    unsigned d = (unsigned)f2bf(w0) | ((unsigned)f2bf(w1) << 16);

    int kk    = k >> 5;
    int kin   = k & 31;
    int kgrp  = (kin >> 3) & 1;
    int v     = ((kin & 7) >> 1) | ((kin >> 4) << 2);
    int lane  = (kgrp << 4) | (n & 15);
    int ntile = n >> 4;
    Bpack[((((size_t)s * 8 + ntile) * 4 + kk) * 32 + lane) * 8 + v] = d;
}

// ---------------------------------------------------------------------------
// 3. Per-(dst,relation) degree count, then invert in place
// ---------------------------------------------------------------------------
__global__ void deg_count_kernel(const int* __restrict__ dst,
                                 const int* __restrict__ attr,
                                 float* __restrict__ deg) {
    int e = blockIdx.x * blockDim.x + threadIdx.x;
    if (e < EE) atomicAdd(&deg[(size_t)dst[e] * RR + attr[e]], 1.0f);
}
__global__ void inv_deg_kernel(float* __restrict__ deg) {
    int i = blockIdx.x * blockDim.x + threadIdx.x;
    if (i < NN * RR) deg[i] = 1.0f / fmaxf(deg[i], 1.0f);
}

// ---------------------------------------------------------------------------
// 4. WMMA GEMM: xw[N, 9*128] = hb[N,128] @ W[9][128,128]  (bf16 -> f32 acc)
//    grid: N/16 blocks of 256 (8 waves). Each wave owns one 16-column tile
//    across ALL 9 slices: A fragment loaded once per k-step (2 x b128),
//    then 9 WMMAs against pre-packed B fragments (2 x b128 each).
// ---------------------------------------------------------------------------
__global__ void gemm_bf16_wmma(const unsigned short* __restrict__ Abf,
                               const unsigned* __restrict__ Bpack,
                               unsigned short* __restrict__ Cbf) {
    const int lane = threadIdx.x & 31;
    const int wave = threadIdx.x >> 5;       // n-tile 0..7
    const int m0 = blockIdx.x * 16;
    const int mrow = m0 + (lane & 15);
    const int khalfA = lane >> 4;            // lanes 16-31 hold upper K octet

    v8f acc[SLICES];
    #pragma unroll
    for (int s = 0; s < SLICES; ++s)
        acc[s] = (v8f){0.f, 0.f, 0.f, 0.f, 0.f, 0.f, 0.f, 0.f};

    const unsigned short* arow = Abf + (size_t)mrow * HH;

    #pragma unroll
    for (int kk = 0; kk < 4; ++kk) {
        const int k0 = kk * 32;
        Frag fa;
        // A fragment (ISA 7.12.2, 16-bit A 16x32): v0..3 <- k0+khalf*8+{0,2,4,6},
        // v4..7 <- k0+16+khalf*8+{0,2,4,6}: two contiguous 16B runs.
        fa.q[0] = *(const uint4*)(arow + k0 + khalfA * 8);
        fa.q[1] = *(const uint4*)(arow + k0 + 16 + khalfA * 8);

        #pragma unroll
        for (int s = 0; s < SLICES; ++s) {
            const unsigned* bp =
                Bpack + ((((size_t)s * 8 + wave) * 4 + kk) * 32 + lane) * 8;
            Frag fb;
            fb.q[0] = *(const uint4*)(bp);
            fb.q[1] = *(const uint4*)(bp + 4);
            acc[s] = __builtin_amdgcn_wmma_f32_16x16x32_bf16(
                         false, fa.v, false, fb.v, (short)0, acc[s], false, false);
        }
    }

    // C/D layout: lane -> N=lane&15, VGPR v -> M = v + 8*(lane>>4)
    const int mloc = ((lane >> 4) << 3);
    const int nloc = lane & 15;
    #pragma unroll
    for (int s = 0; s < SLICES; ++s) {
        #pragma unroll
        for (int v = 0; v < 8; ++v) {
            Cbf[(size_t)(m0 + mloc + v) * (SLICES * HH) + s * HH + wave * 16 + nloc]
                = f2bf(acc[s][v]);
        }
    }
}

// ---------------------------------------------------------------------------
// 5. Edge scatter: one wave per edge; gather bf16 row of xw[src] for its
//    relation, scale by 1/deg(dst,r), atomic-add into accum[dst].
// ---------------------------------------------------------------------------
__global__ void edge_scatter_kernel(const int* __restrict__ src,
                                    const int* __restrict__ dst,
                                    const int* __restrict__ attr,
                                    const unsigned short* __restrict__ xw,
                                    const float* __restrict__ invdeg,
                                    float* __restrict__ accum) {
    int e = blockIdx.x * 8 + (threadIdx.x >> 5);   // E divisible by 8
    int lane = threadIdx.x & 31;
    int sn = src[e], dn = dst[e], r = attr[e];
    float w = invdeg[(size_t)dn * RR + r];
    const uint2 pk = *(const uint2*)(xw + (size_t)sn * (SLICES * HH)
                                        + (size_t)(1 + r) * HH + lane * 4);
    float f0 = bf2f((unsigned short)(pk.x & 0xffffu));
    float f1 = bf2f((unsigned short)(pk.x >> 16));
    float f2 = bf2f((unsigned short)(pk.y & 0xffffu));
    float f3 = bf2f((unsigned short)(pk.y >> 16));
    float* ap = accum + (size_t)dn * HH + lane * 4;
    atomicAdd(ap + 0, w * f0);
    atomicAdd(ap + 1, w * f1);
    atomicAdd(ap + 2, w * f2);
    atomicAdd(ap + 3, w * f3);
}

// ---------------------------------------------------------------------------
// 6. Finalize layer: h = relu(root_part + bias + aggregated) -> bf16
// ---------------------------------------------------------------------------
__global__ void finalize_kernel(const unsigned short* __restrict__ xw,
                                const float* __restrict__ accum,
                                const float* __restrict__ bias, int l,
                                unsigned short* __restrict__ hb) {
    int i = blockIdx.x * blockDim.x + threadIdx.x;   // < N*H
    int c = i & (HH - 1);
    int n = i >> 7;
    float v = bf2f(xw[(size_t)n * (SLICES * HH) + c]) + bias[l * HH + c] + accum[i];
    hb[i] = f2bf(fmaxf(v, 0.f));
}

// ---------------------------------------------------------------------------
// 7. Global mean pool per graph (atomic)
// ---------------------------------------------------------------------------
__global__ void pool_kernel(const unsigned short* __restrict__ hb,
                            const int* __restrict__ batch,
                            float* __restrict__ pool, float* __restrict__ cnt) {
    int i = blockIdx.x * blockDim.x + threadIdx.x;   // < N*H
    int n = i >> 7, c = i & (HH - 1);
    int g = batch[n];
    atomicAdd(&pool[(size_t)g * HH + c], bf2f(hb[i]));
    if (c == 0) atomicAdd(&cnt[g], 1.0f);
}

// ---------------------------------------------------------------------------
// 8. Head: BN2 -> mean over graphs -> relu -> fc1 -> relu -> fc2 -> log_softmax
// ---------------------------------------------------------------------------
__global__ void head_kernel(const float* __restrict__ pool, const float* __restrict__ cnt,
                            const float* __restrict__ g2, const float* __restrict__ b2,
                            const float* __restrict__ m2, const float* __restrict__ v2,
                            const float* __restrict__ fc1w, const float* __restrict__ fc1b,
                            const float* __restrict__ fc2w, const float* __restrict__ fc2b,
                            float* __restrict__ out) {
    __shared__ float sv[HH];
    __shared__ float su[HH];
    __shared__ float slg[CC];
    int t = threadIdx.x;   // 128 threads
    float acc = 0.f;
    for (int gg = 0; gg < GG; ++gg) {
        float gm = pool[gg * HH + t] / fmaxf(cnt[gg], 1.0f);
        acc += (gm - m2[t]) * rsqrtf(v2[t] + EPSF) * g2[t] + b2[t];
    }
    sv[t] = fmaxf(acc * (1.0f / (float)GG), 0.f);
    __syncthreads();
    float u = fc1b[t];
    for (int c = 0; c < HH; ++c) u += sv[c] * fc1w[c * HH + t];
    su[t] = fmaxf(u, 0.f);
    __syncthreads();
    if (t < CC) {
        float lg = fc2b[t];
        for (int o = 0; o < HH; ++o) lg += su[o] * fc2w[o * CC + t];
        slg[t] = lg;
    }
    __syncthreads();
    if (t == 0) {
        float mx = slg[0];
        for (int k = 1; k < CC; ++k) mx = fmaxf(mx, slg[k]);
        float se = 0.f;
        for (int k = 0; k < CC; ++k) se += __expf(slg[k] - mx);
        float lse = __logf(se);
        for (int k = 0; k < CC; ++k) out[k] = slg[k] - mx - lse;
    }
}

// ---------------------------------------------------------------------------
// Host orchestration
// ---------------------------------------------------------------------------
extern "C" void kernel_launch(void* const* d_in, const int* in_sizes, int n_in,
                              void* d_out, int out_size, void* d_ws, size_t ws_size,
                              hipStream_t stream) {
    (void)in_sizes; (void)n_in; (void)out_size; (void)ws_size;

    const float* x        = (const float*)d_in[0];
    const int*   eidx     = (const int*)  d_in[1];   // [2,E]
    const int*   eattr    = (const int*)  d_in[2];
    const int*   batch    = (const int*)  d_in[3];
    const float* bn1_g    = (const float*)d_in[4];
    const float* bn1_b    = (const float*)d_in[5];
    const float* bn1_m    = (const float*)d_in[6];
    const float* bn1_v    = (const float*)d_in[7];
    const float* basis    = (const float*)d_in[8];
    const float* comp     = (const float*)d_in[9];
    const float* root     = (const float*)d_in[10];
    const float* bias     = (const float*)d_in[11];
    const float* bn2_g    = (const float*)d_in[12];
    const float* bn2_b    = (const float*)d_in[13];
    const float* bn2_m    = (const float*)d_in[14];
    const float* bn2_v    = (const float*)d_in[15];
    const float* fc1_w    = (const float*)d_in[16];
    const float* fc1_b    = (const float*)d_in[17];
    const float* fc2_w    = (const float*)d_in[18];
    const float* fc2_b    = (const float*)d_in[19];
    float* out = (float*)d_out;

    const int* src = eidx;
    const int* dst = eidx + EE;

    // -------- workspace carve-up (all sizes multiples of 256B) --------
    char* ws = (char*)d_ws;
    size_t off = 0;
    unsigned short* hb  = (unsigned short*)(ws + off); off += (size_t)NN * HH * 2;            // 12.8 MB
    unsigned short* xw  = (unsigned short*)(ws + off); off += (size_t)NN * SLICES * HH * 2;   // 115.2 MB
    float* accum        = (float*)(ws + off);          off += (size_t)NN * HH * 4;            // 25.6 MB
    float* invdeg       = (float*)(ws + off);          off += (size_t)NN * RR * 4;            // 1.6 MB
    unsigned* Bpack     = (unsigned*)(ws + off);       off += (size_t)SLICES * HH * HH * 2;   // 294 KB
    float* pool         = (float*)(ws + off);          off += (size_t)GG * HH * 4;
    float* cnt          = (float*)(ws + off);          off += (size_t)GG * 4;

    const int NH_BLOCKS = (NN * HH) / 256;      // 25000

    // BN1 -> bf16 activations
    bn1_kernel<<<NH_BLOCKS, 256, 0, stream>>>(x, bn1_g, bn1_b, bn1_m, bn1_v, hb);

    // degree table (topology-invariant across layers)
    hipMemsetAsync(invdeg, 0, (size_t)NN * RR * 4, stream);
    deg_count_kernel<<<(EE + 255) / 256, 256, 0, stream>>>(dst, eattr, invdeg);
    inv_deg_kernel<<<(NN * RR + 255) / 256, 256, 0, stream>>>(invdeg);

    for (int l = 0; l < LL; ++l) {
        // per-relation weights, packed into WMMA B-fragment layout (bf16)
        wcombine_pack_kernel<<<(SLICES * 64 * HH) / 256, 256, 0, stream>>>(
            basis, comp, root, l, Bpack);
        // xw = h @ [root | W_0..W_7]  via bf16 WMMA (9 slices per wave)
        gemm_bf16_wmma<<<NN / 16, 256, 0, stream>>>(hb, Bpack, xw);
        // mean-per-relation aggregation
        hipMemsetAsync(accum, 0, (size_t)NN * HH * 4, stream);
        edge_scatter_kernel<<<EE / 8, 256, 0, stream>>>(src, dst, eattr, xw, invdeg, accum);
        // h = relu(root + bias + agg)
        finalize_kernel<<<NH_BLOCKS, 256, 0, stream>>>(xw, accum, bias, l, hb);
    }

    // mean pool per graph, then head
    hipMemsetAsync(pool, 0, (size_t)(GG * HH + GG) * 4, stream);
    pool_kernel<<<NH_BLOCKS, 256, 0, stream>>>(hb, batch, pool, cnt);
    head_kernel<<<1, HH, 0, stream>>>(pool, cnt, bn2_g, bn2_b, bn2_m, bn2_v,
                                      fc1_w, fc1_b, fc2_w, fc2_b, out);
}
